// Decoder_Network_55868934586633
// MI455X (gfx1250) — compile-verified, matrix operand order checked by reading, not run on hardware
//
#include <hip/hip_runtime.h>
#include <hip/hip_bf16.h>

// ---------------------------------------------------------------------------
// Decoder network for MI455X (gfx1250, wave32, WMMA).
//   1) convert x and GEMM weights to f16 (v_wmma_f32_16x16x32_f16, f32 accum)
//   2) FPS: one persistent 1024-thread workgroup per batch, dists in regs
//   3) per branch: 4x [WMMA GEMM -> col-stats -> BN+ReLU->f16]
//      AXform: 2x [GEMM->BN->f16], 1x GEMM+bias, column softmax (written
//      transposed), base transpose, per-batch WMMA GEMM (w·base), 3x64 proj.
// GEMM wave tile is 32(M)x64(N): 8 WMMAs per 32-k step off 6 fragment loads.
// ---------------------------------------------------------------------------

typedef __attribute__((ext_vector_type(16))) _Float16 v16h;
typedef __attribute__((ext_vector_type(8)))  _Float16 v8h;
typedef __attribute__((ext_vector_type(8)))  float    v8f;

#define NBRANCH 8
#define BATCH   32
#define LSEQ    512
#define ROWS    (BATCH * LSEQ)   // 16384
#define K1      64
#define NPART   32768
#define NPOINT  1024
#define BN_EPS  1e-5f

// -------------------------- helpers ---------------------------------------

__device__ __forceinline__ v8f zero8() {
  v8f z;
#pragma unroll
  for (int i = 0; i < 8; ++i) z[i] = 0.0f;
  return z;
}

// Load one 16x32 f16 fragment row for this lane (A row-of-X or B row-of-W).
// Layout per ISA 7.12.2: lanes 0-15 hold K=[0..7],[16..23]; lanes 16-31 hold
// K=[8..15],[24..31] of row (lane&15). Two 16B loads -> merged b128 pairs.
__device__ __forceinline__ v16h load_frag(const _Float16* __restrict__ rowPtr,
                                          int k0, int laneHalf) {
  v8h lo = *(const v8h*)(rowPtr + k0 + laneHalf * 8);
  v8h hi = *(const v8h*)(rowPtr + k0 + 16 + laneHalf * 8);
  v16h r;
#pragma unroll
  for (int i = 0; i < 8; ++i) { r[i] = lo[i]; r[8 + i] = hi[i]; }
  return r;
}

// -------------------------- kernels ---------------------------------------

__global__ void convert_f32_f16_kernel(const float* __restrict__ in,
                                       _Float16* __restrict__ out, long n) {
  long i = (long)blockIdx.x * blockDim.x + threadIdx.x;
  if (i < n) out[i] = (_Float16)in[i];
}

// C[M,N] = A[M,K] * W[N,K]^T (+bias). One wave per 32x64 output tile,
// 4 waves / block on consecutive M tiles. grid.z batches A/B/C by strides.
__global__ void gemm_wmma_kernel(const _Float16* __restrict__ A,
                                 const _Float16* __restrict__ Bw,
                                 const float* __restrict__ bias,
                                 float* __restrict__ C,
                                 int M, int N, int K,
                                 long strideAz, long strideBz, long strideCz) {
  const int z = blockIdx.z;
  A  += (long)z * strideAz;
  Bw += (long)z * strideBz;
  C  += (long)z * strideCz;

  const int wave = threadIdx.x >> 5;
  const int lane = threadIdx.x & 31;
  const int laneHalf = lane >> 4;  // 0: lanes 0-15, 1: lanes 16-31
  const int ln = lane & 15;

  const int m0 = (blockIdx.x * 4 + wave) * 32;
  if (m0 >= M) return;
  const int n0 = blockIdx.y * 64;

  const _Float16* rowA0 = A + (long)(m0 + ln) * K;
  const _Float16* rowA1 = A + (long)(m0 + 16 + ln) * K;
  const _Float16* rowB0 = Bw + (long)(n0 +  0 + ln) * K;
  const _Float16* rowB1 = Bw + (long)(n0 + 16 + ln) * K;
  const _Float16* rowB2 = Bw + (long)(n0 + 32 + ln) * K;
  const _Float16* rowB3 = Bw + (long)(n0 + 48 + ln) * K;

  v8f c00 = zero8(), c01 = zero8(), c02 = zero8(), c03 = zero8();
  v8f c10 = zero8(), c11 = zero8(), c12 = zero8(), c13 = zero8();

  for (int k0 = 0; k0 < K; k0 += 32) {
    v16h a0 = load_frag(rowA0, k0, laneHalf);
    v16h a1 = load_frag(rowA1, k0, laneHalf);
    v16h b0 = load_frag(rowB0, k0, laneHalf);
    v16h b1 = load_frag(rowB1, k0, laneHalf);
    v16h b2 = load_frag(rowB2, k0, laneHalf);
    v16h b3 = load_frag(rowB3, k0, laneHalf);
    c00 = __builtin_amdgcn_wmma_f32_16x16x32_f16(false, a0, false, b0, (short)0, c00, false, false);
    c10 = __builtin_amdgcn_wmma_f32_16x16x32_f16(false, a1, false, b0, (short)0, c10, false, false);
    c01 = __builtin_amdgcn_wmma_f32_16x16x32_f16(false, a0, false, b1, (short)0, c01, false, false);
    c11 = __builtin_amdgcn_wmma_f32_16x16x32_f16(false, a1, false, b1, (short)0, c11, false, false);
    c02 = __builtin_amdgcn_wmma_f32_16x16x32_f16(false, a0, false, b2, (short)0, c02, false, false);
    c12 = __builtin_amdgcn_wmma_f32_16x16x32_f16(false, a1, false, b2, (short)0, c12, false, false);
    c03 = __builtin_amdgcn_wmma_f32_16x16x32_f16(false, a0, false, b3, (short)0, c03, false, false);
    c13 = __builtin_amdgcn_wmma_f32_16x16x32_f16(false, a1, false, b3, (short)0, c13, false, false);
  }

  float bv0 = 0.f, bv1 = 0.f, bv2 = 0.f, bv3 = 0.f;
  if (bias) {
    bv0 = bias[n0 + ln];      bv1 = bias[n0 + 16 + ln];
    bv2 = bias[n0 + 32 + ln]; bv3 = bias[n0 + 48 + ln];
  }
  // C/D layout: VGPR r -> M = mtile + r + laneHalf*8, N = n0 + 16*j + ln
  const int mbase0 = m0 + laneHalf * 8;
#pragma unroll
  for (int r = 0; r < 8; ++r) {
    float* crow = C + (long)(mbase0 + r) * N;
    crow[n0 + ln]      = c00[r] + bv0;
    crow[n0 + 16 + ln] = c01[r] + bv1;
    crow[n0 + 32 + ln] = c02[r] + bv2;
    crow[n0 + 48 + ln] = c03[r] + bv3;
  }
  const int mbase1 = m0 + 16 + laneHalf * 8;
#pragma unroll
  for (int r = 0; r < 8; ++r) {
    float* crow = C + (long)(mbase1 + r) * N;
    crow[n0 + ln]      = c10[r] + bv0;
    crow[n0 + 16 + ln] = c11[r] + bv1;
    crow[n0 + 32 + ln] = c12[r] + bv2;
    crow[n0 + 48 + ln] = c13[r] + bv3;
  }
}

// Per-column mean / rsqrt(var+eps) over R rows. 32 channels per block (one
// per lane -> coalesced), 8 waves stride rows, LDS cross-wave reduce.
__global__ void colstats_kernel(const float* __restrict__ H, int R, int N,
                                float* __restrict__ mean,
                                float* __restrict__ rinv) {
  const int lane = threadIdx.x & 31;
  const int w = threadIdx.x >> 5;          // 0..7
  const int ch = blockIdx.x * 32 + lane;
  float s = 0.f, s2 = 0.f;
  if (ch < N) {
    for (int r = w; r < R; r += 8) {
      float v = H[(long)r * N + ch];
      s += v; s2 += v * v;
    }
  }
  __shared__ float sh[8][32], sh2[8][32];
  sh[w][lane] = s; sh2[w][lane] = s2;
  __syncthreads();
  if (w == 0) {
#pragma unroll
    for (int i = 1; i < 8; ++i) { s += sh[i][lane]; s2 += sh2[i][lane]; }
    if (ch < N) {
      float m = s / (float)R;
      float var = s2 / (float)R - m * m;
      mean[ch] = m;
      rinv[ch] = rsqrtf(var + BN_EPS);
    }
  }
}

// BN (training stats) + ReLU, emitting f16 activations for the next GEMM.
__global__ void bn_relu_f16_kernel(const float* __restrict__ H, long total, int N,
                                   const float* __restrict__ mean,
                                   const float* __restrict__ rinv,
                                   const float* __restrict__ gamma,
                                   const float* __restrict__ beta,
                                   _Float16* __restrict__ out) {
  long i = (long)blockIdx.x * blockDim.x + threadIdx.x;
  if (i >= total) return;
  int ch = (int)(i % N);
  float v = (H[i] - mean[ch]) * rinv[ch] * gamma[ch] + beta[ch];
  out[i] = (_Float16)fmaxf(v, 0.0f);
}

// Column softmax of t3 over L (per (b,m)); writes w transposed: [b][m][l] f16.
__global__ void softmax_col_kernel(const float* __restrict__ T,  // [B*L,128]
                                   _Float16* __restrict__ Wt) {  // [B,128,L]
  const int bm = blockIdx.x;           // b*128 + m
  const int b = bm >> 7, m = bm & 127;
  const int l = threadIdx.x;           // 0..511
  float v = T[((long)b * LSEQ + l) * 128 + m];
  __shared__ float sh[LSEQ];
  sh[l] = v; __syncthreads();
  for (int off = LSEQ / 2; off > 0; off >>= 1) {
    if (l < off) sh[l] = fmaxf(sh[l], sh[l + off]);
    __syncthreads();
  }
  float mx = sh[0];
  __syncthreads();
  float e = __expf(v - mx);
  sh[l] = e; __syncthreads();
  for (int off = LSEQ / 2; off > 0; off >>= 1) {
    if (l < off) sh[l] += sh[l + off];
    __syncthreads();
  }
  float sum = sh[0];
  Wt[(long)bm * LSEQ + l] = (_Float16)(e / sum);
}

// base [B*L,64] f16 -> baseT [B,64,L] f16 (rows become K-contiguous for WMMA B)
__global__ void transpose_base_kernel(const _Float16* __restrict__ in,
                                      _Float16* __restrict__ out, long total) {
  long i = (long)blockIdx.x * blockDim.x + threadIdx.x;
  if (i >= total) return;
  int l = (int)(i % LSEQ);
  int k = (int)((i / LSEQ) % K1);
  int b = (int)(i / ((long)LSEQ * K1));
  out[i] = in[((long)b * LSEQ + l) * K1 + k];
}

// out[b, branch*128+m, o] = aW4[o,:] . pts[b,m,:] + ab4[o]
__global__ void final_proj_kernel(const float* __restrict__ pts,  // [B,128,64]
                                  const float* __restrict__ aW4,  // [3,64]
                                  const float* __restrict__ ab4,  // [3]
                                  float* __restrict__ outp, int branch) {
  int t = blockIdx.x * blockDim.x + threadIdx.x;  // B*128*3
  if (t >= BATCH * 128 * 3) return;
  int o = t % 3, m = (t / 3) % 128, b = t / (3 * 128);
  const float* p = pts + ((long)b * 128 + m) * K1;
  const float* w = aW4 + o * K1;
  float acc = ab4[o];
#pragma unroll 8
  for (int k = 0; k < K1; ++k) acc += w[k] * p[k];
  outp[((long)b * 2048 + branch * 128 + m) * 3 + o] = acc;
}

// Farthest point sampling: one 1024-thread workgroup per batch; per-thread
// min-distances in registers; LDS argmax tree each of 1024 sequential steps.
// Writes selected coords straight into out rows [1024, 2048).
__global__ void __launch_bounds__(1024)
fps_kernel(const float* __restrict__ xp, float* __restrict__ outp) {
  const int b = blockIdx.x;
  const float* base = xp + (long)b * NPART * 3;
  const int tid = threadIdx.x;                 // 0..1023
  const int PPT = NPART / 1024;                // 32 points per thread

  float dist[NPART / 1024];
#pragma unroll
  for (int i = 0; i < PPT; ++i) dist[i] = 1e10f;

  __shared__ float sval[1024];
  __shared__ int   sidx[1024];
  __shared__ float cpt[3];

  int far = 0;
  for (int j = 0; j < NPOINT; ++j) {
    if (tid == 0) {
      float x = base[far * 3 + 0], y = base[far * 3 + 1], z = base[far * 3 + 2];
      cpt[0] = x; cpt[1] = y; cpt[2] = z;
      float* orow = outp + ((long)b * 2048 + NPOINT + j) * 3;
      orow[0] = x; orow[1] = y; orow[2] = z;
    }
    __syncthreads();
    const float cx = cpt[0], cy = cpt[1], cz = cpt[2];
    float best = -1.0f; int bi = 0x7fffffff;
#pragma unroll
    for (int i = 0; i < PPT; ++i) {
      const int p = i * 1024 + tid;            // coalesced
      float dx = base[p * 3 + 0] - cx;
      float dy = base[p * 3 + 1] - cy;
      float dz = base[p * 3 + 2] - cz;
      float d = dx * dx + dy * dy + dz * dz;
      float nd = fminf(dist[i], d);
      dist[i] = nd;
      if (nd > best) { best = nd; bi = p; }
    }
    sval[tid] = best; sidx[tid] = bi;
    __syncthreads();
    for (int off = 512; off > 0; off >>= 1) {
      if (tid < off) {
        float v2 = sval[tid + off];
        if (v2 > sval[tid] || (v2 == sval[tid] && sidx[tid + off] < sidx[tid])) {
          sval[tid] = v2; sidx[tid] = sidx[tid + off];
        }
      }
      __syncthreads();
    }
    far = sidx[0];
  }
}

// -------------------------- host launcher ----------------------------------

extern "C" void kernel_launch(void* const* d_in, const int* in_sizes, int n_in,
                              void* d_out, int out_size, void* d_ws, size_t ws_size,
                              hipStream_t stream) {
  (void)in_sizes; (void)n_in; (void)out_size; (void)ws_size;

  const float* x      = (const float*)d_in[0];   // [32,512,256]
  const float* x_part = (const float*)d_in[1];   // [32,32768,3]
  const float* mW[4]  = { (const float*)d_in[2],  (const float*)d_in[6],
                          (const float*)d_in[10], (const float*)d_in[14] };
  const float* mb[4]  = { (const float*)d_in[3],  (const float*)d_in[7],
                          (const float*)d_in[11], (const float*)d_in[15] };
  const float* mg[4]  = { (const float*)d_in[4],  (const float*)d_in[8],
                          (const float*)d_in[12], (const float*)d_in[16] };
  const float* mB[4]  = { (const float*)d_in[5],  (const float*)d_in[9],
                          (const float*)d_in[13], (const float*)d_in[17] };
  const float* aW1 = (const float*)d_in[18]; const float* ab1 = (const float*)d_in[19];
  const float* ag1 = (const float*)d_in[20]; const float* aB1 = (const float*)d_in[21];
  const float* aW2 = (const float*)d_in[22]; const float* ab2 = (const float*)d_in[23];
  const float* ag2 = (const float*)d_in[24]; const float* aB2 = (const float*)d_in[25];
  const float* aW3 = (const float*)d_in[26]; const float* ab3 = (const float*)d_in[27];
  const float* aW4 = (const float*)d_in[28]; const float* ab4 = (const float*)d_in[29];
  float* out = (float*)d_out;                    // [32,2048,3]

  // ---- workspace carve-out (256B aligned) ----
  size_t off = 0;
  auto take = [&](size_t bytes) -> char* {
    char* p = (char*)d_ws + off;
    off += (bytes + 255) & ~(size_t)255;
    return p;
  };
  _Float16* x16     = (_Float16*)take((size_t)ROWS * 256 * 2);
  _Float16* w_mW1   = (_Float16*)take((size_t)NBRANCH * 512 * 256 * 2);
  _Float16* w_mW2   = (_Float16*)take((size_t)NBRANCH * 512 * 512 * 2);
  _Float16* w_mW3   = (_Float16*)take((size_t)NBRANCH * 512 * 512 * 2);
  _Float16* w_mW4   = (_Float16*)take((size_t)NBRANCH * K1 * 512 * 2);
  _Float16* w_aW1   = (_Float16*)take((size_t)NBRANCH * 64 * K1 * 2);
  _Float16* w_aW2   = (_Float16*)take((size_t)NBRANCH * 128 * 64 * 2);
  _Float16* w_aW3   = (_Float16*)take((size_t)NBRANCH * 128 * 128 * 2);
  float*    buf1    = (float*)   take((size_t)ROWS * 512 * 4);   // GEMM f32 out
  _Float16* buf2    = (_Float16*)take((size_t)ROWS * 512 * 2);   // f16 acts
  _Float16* base16  = (_Float16*)take((size_t)ROWS * K1 * 2);    // layer4 acts
  _Float16* baseT   = (_Float16*)take((size_t)BATCH * K1 * LSEQ * 2);
  _Float16* wT      = (_Float16*)take((size_t)BATCH * 128 * LSEQ * 2);
  float*    pts     = (float*)   take((size_t)BATCH * 128 * K1 * 4);
  float*    meanb   = (float*)   take(512 * 4);
  float*    rinvb   = (float*)   take(512 * 4);

  auto cvt = [&](const float* src, _Float16* dst, long n) {
    convert_f32_f16_kernel<<<dim3((unsigned)((n + 255) / 256)), dim3(256), 0, stream>>>(src, dst, n);
  };
  // one-time conversions
  cvt(x, x16, (long)ROWS * 256);
  cvt(mW[0], w_mW1, (long)NBRANCH * 512 * 256);
  cvt(mW[1], w_mW2, (long)NBRANCH * 512 * 512);
  cvt(mW[2], w_mW3, (long)NBRANCH * 512 * 512);
  cvt(mW[3], w_mW4, (long)NBRANCH * K1 * 512);
  cvt(aW1, w_aW1, (long)NBRANCH * 64 * K1);
  cvt(aW2, w_aW2, (long)NBRANCH * 128 * 64);
  cvt(aW3, w_aW3, (long)NBRANCH * 128 * 128);

  // FPS fills out rows [1024,2048) — independent of GEMM chain
  fps_kernel<<<dim3(BATCH), dim3(1024), 0, stream>>>(x_part, out);

  auto gemm = [&](const _Float16* A, const _Float16* Bw, const float* bias,
                  float* C, int M, int N, int K,
                  long sAz, long sBz, long sCz, int Z) {
    dim3 grid((unsigned)((M / 32 + 3) / 4), (unsigned)(N / 64), (unsigned)Z);
    gemm_wmma_kernel<<<grid, dim3(128), 0, stream>>>(A, Bw, bias, C, M, N, K, sAz, sBz, sCz);
  };
  auto stats = [&](const float* H, int N) {
    colstats_kernel<<<dim3((unsigned)((N + 31) / 32)), dim3(256), 0, stream>>>(
        H, ROWS, N, meanb, rinvb);
  };
  auto bnrelu = [&](const float* H, int N, const float* g, const float* be, _Float16* o) {
    long total = (long)ROWS * N;
    bn_relu_f16_kernel<<<dim3((unsigned)((total + 255) / 256)), dim3(256), 0, stream>>>(
        H, total, N, meanb, rinvb, g, be, o);
  };

  const int mwN[4] = {512, 512, 512, K1};
  const int mwK[4] = {256, 512, 512, 512};
  _Float16* mw16[4] = {w_mW1, w_mW2, w_mW3, w_mW4};

  for (int br = 0; br < NBRANCH; ++br) {
    // ---- MappingNet: 4 x (GEMM + BN + ReLU) ----
    const _Float16* Ain = x16;
    for (int L = 0; L < 4; ++L) {
      const int N = mwN[L], K = mwK[L];
      const _Float16* W = mw16[L] + (long)br * N * K;
      gemm(Ain, W, mb[L] + (long)br * N, buf1, ROWS, N, K, 0, 0, 0, 1);
      stats(buf1, N);
      _Float16* actOut = (L == 3) ? base16 : buf2;
      bnrelu(buf1, N, mg[L] + (long)br * N, mB[L] + (long)br * N, actOut);
      Ain = actOut;
    }
    // ---- AXform conv1 (64), conv2 (128): GEMM + BN + ReLU ----
    gemm(base16, w_aW1 + (long)br * 64 * K1, ab1 + (long)br * 64, buf1, ROWS, 64, K1, 0, 0, 0, 1);
    stats(buf1, 64);
    bnrelu(buf1, 64, ag1 + (long)br * 64, aB1 + (long)br * 64, buf2);

    gemm(buf2, w_aW2 + (long)br * 128 * 64, ab2 + (long)br * 128, buf1, ROWS, 128, 64, 0, 0, 0, 1);
    stats(buf1, 128);
    bnrelu(buf1, 128, ag2 + (long)br * 128, aB2 + (long)br * 128, buf2);

    // ---- conv3: GEMM + bias only (logits t3 into buf1, f32) ----
    gemm(buf2, w_aW3 + (long)br * 128 * 128, ab3 + (long)br * 128, buf1, ROWS, 128, 128, 0, 0, 0, 1);

    // ---- softmax over L per (b,m), written transposed for the pts GEMM ----
    softmax_col_kernel<<<dim3(BATCH * 128), dim3(LSEQ), 0, stream>>>(buf1, wT);

    // ---- transpose base -> [b][k][l] so WMMA B rows are contiguous ----
    {
      long total = (long)BATCH * K1 * LSEQ;
      transpose_base_kernel<<<dim3((unsigned)((total + 255) / 256)), dim3(256), 0, stream>>>(
          base16, baseT, total);
    }

    // ---- pts[b] = w[b] (128xL) * base[b] (LxK1), per-batch WMMA GEMM ----
    gemm(wT, baseT, nullptr, pts, 128, K1, LSEQ,
         (long)128 * LSEQ, (long)K1 * LSEQ, (long)128 * K1, BATCH);

    // ---- final 3x64 projection straight into out rows [br*128, br*128+128) --
    final_proj_kernel<<<dim3((BATCH * 128 * 3 + 255) / 256), dim3(256), 0, stream>>>(
        pts, aW4 + (long)br * 3 * K1, ab4 + (long)br * 3, out, br);
  }
}
